// CoconutLearnableLora_8065948582558
// MI455X (gfx1250) — matrix operand order repaired and checked by reading, not compiled, not run on hardware
//
#include <hip/hip_runtime.h>
#include <hip/hip_bf16.h>

// ---------- types ----------
typedef __attribute__((ext_vector_type(16))) __bf16 v16bf;
typedef __attribute__((ext_vector_type(8)))  __bf16 v8bf;
typedef __attribute__((ext_vector_type(8)))  float  v8f;
typedef unsigned int v4u __attribute__((ext_vector_type(4)));
typedef int          v4i __attribute__((ext_vector_type(4)));
typedef int          v8i __attribute__((ext_vector_type(8)));

#define DMODEL 1024
#define TSEQ   1024
#define NHEAD  16
#define DHEAD  64
#define DFF_   4096
#define VOCAB  32000

__device__ __forceinline__ v8f wmma_bf16(v16bf a, v16bf b, v8f c) {
  return __builtin_amdgcn_wmma_f32_16x16x32_bf16(false, a, false, b, (short)0, c,
                                                 false, false);
}

__device__ __forceinline__ v16bf mk16(v8bf lo, v8bf hi) {
  v16bf r;
#pragma unroll
  for (int i = 0; i < 8; ++i) { r[i] = lo[i]; r[i + 8] = hi[i]; }
  return r;
}

__device__ __forceinline__ float rmax16(float v) {
#pragma unroll
  for (int m = 1; m <= 8; m <<= 1) v = fmaxf(v, __shfl_xor(v, m, 32));
  return v;
}
__device__ __forceinline__ float rsum16(float v) {
#pragma unroll
  for (int m = 1; m <= 8; m <<= 1) v += __shfl_xor(v, m, 32);
  return v;
}

__device__ __forceinline__ float gelu_tanh(float x) {
  float x3 = x * x * x;
  return 0.5f * x * (1.f + tanhf(0.7978845608f * (x + 0.044715f * x3)));
}

// ---------- TDM: async 2D bf16 tile load (global -> LDS), TENSORcnt tracked ----
// Descriptor packing per CDNA5 ISA.
__device__ __forceinline__ void tdm_load_2d_bf16(const __bf16* gptr, void* lds,
                                                 unsigned tensor_d0, unsigned tensor_d1,
                                                 unsigned tile_d0, unsigned tile_d1,
                                                 unsigned long long stride0_elems) {
  unsigned long long ga = (unsigned long long)(size_t)gptr;
  unsigned int laddr = (unsigned int)(size_t)lds;  // generic LDS ptr low 32b = offset
  v4u g0;
  g0[0] = 1u;                                        // count=1, user descriptor
  g0[1] = laddr;                                     // lds_addr
  g0[2] = (unsigned int)(ga & 0xFFFFFFFFu);          // global_addr[31:0]
  g0[3] = (unsigned int)((ga >> 32) & 0x1FFFFFFu) |  // global_addr[56:32]
          (2u << 30);                                // type=2 (image)
  v8i g1;
  g1[0] = (int)(1u << 16);                           // data_size=2B; mask=0
  g1[1] = (int)((tensor_d0 & 0xFFFFu) << 16);        // tensor_dim0[15:0]
  g1[2] = (int)((tensor_d0 >> 16) | ((tensor_d1 & 0xFFFFu) << 16));
  g1[3] = (int)((tensor_d1 >> 16) | (tile_d0 << 16)); // tile_dim0
  g1[4] = (int)(tile_d1);                            // tile_dim1 (tile_dim2=0)
  g1[5] = (int)(stride0_elems & 0xFFFFFFFFull);      // tensor_dim0_stride[31:0]
  g1[6] = (int)((stride0_elems >> 32) & 0xFFFFull);  // stride[47:32]; dim1_stride=0
  g1[7] = 0;
  v4i gz = {0, 0, 0, 0};
#if defined(__clang_major__) && (__clang_major__ >= 23)
  v8i gz8 = {0, 0, 0, 0, 0, 0, 0, 0};
  __builtin_amdgcn_tensor_load_to_lds(g0, g1, gz, gz, gz8, 0);
#else
  __builtin_amdgcn_tensor_load_to_lds(g0, g1, gz, gz, 0);
#endif
}

// ---------- small kernels ----------
// LoRA merge + write TRANSPOSED bf16:  WmT[j*Dd + i] = bf16(W[i,j] + 2*A@B)
__global__ void k_merge_t(const float* __restrict__ W, const float* __restrict__ A,
                          const float* __restrict__ Bl, __bf16* __restrict__ WmT,
                          int Dd, int N, long total) {
  long idx = (long)blockIdx.x * 256 + threadIdx.x;
  if (idx >= total) return;
  int i = (int)(idx / N), j = (int)(idx % N);
  float s = 0.f;
#pragma unroll
  for (int r = 0; r < 8; ++r) s += A[i * 8 + r] * Bl[r * N + j];
  WmT[(size_t)j * Dd + i] = (__bf16)(W[idx] + 2.0f * s);  // ALPHA/R = 2
}

// plain transpose+convert: W[K,N] fp32 -> Wt[N,K] bf16
__global__ void k_convT(const float* __restrict__ W, __bf16* __restrict__ Wt,
                        int K, int N, long total) {
  long idx = (long)blockIdx.x * 256 + threadIdx.x;
  if (idx >= total) return;
  int k = (int)(idx / N), n = (int)(idx % N);
  Wt[(size_t)n * K + k] = (__bf16)W[idx];
}

__global__ void k_embed(const int* __restrict__ ids, const float* __restrict__ emb,
                        float* __restrict__ out) {
  int bt = blockIdx.x;
  int id = ids[bt];
  const float* src = emb + (size_t)id * DMODEL;
  float* dst = out + (size_t)bt * DMODEL;
  for (int d = threadIdx.x; d < DMODEL; d += 256) dst[d] = src[d];
}

__global__ void k_addpos(const float* __restrict__ embeds, const float* __restrict__ Wpos,
                         float* __restrict__ h, int S) {
  int r = blockIdx.x;  // r = b*S + s
  int b = r / S, s = r % S;
  const float* e = embeds + ((size_t)(b * TSEQ + s)) * DMODEL;
  const float* p = Wpos + (size_t)s * DMODEL;
  float* o = h + (size_t)r * DMODEL;
  for (int d = threadIdx.x; d < DMODEL; d += 256) o[d] = e[d] + p[d];
}

__global__ void k_ln(const float* __restrict__ x, const float* __restrict__ g,
                     const float* __restrict__ be, float* __restrict__ y) {
  __shared__ float red[256];
  int r = blockIdx.x, tid = threadIdx.x;
  const float* xr = x + (size_t)r * DMODEL;
  float s = 0.f;
  for (int d = tid; d < DMODEL; d += 256) s += xr[d];
  red[tid] = s; __syncthreads();
  for (int o = 128; o > 0; o >>= 1) { if (tid < o) red[tid] += red[tid + o]; __syncthreads(); }
  float mean = red[0] * (1.0f / DMODEL); __syncthreads();
  float vs = 0.f;
  for (int d = tid; d < DMODEL; d += 256) { float t = xr[d] - mean; vs += t * t; }
  red[tid] = vs; __syncthreads();
  for (int o = 128; o > 0; o >>= 1) { if (tid < o) red[tid] += red[tid + o]; __syncthreads(); }
  float rst = rsqrtf(red[0] * (1.0f / DMODEL) + 1e-5f);
  float* yr = y + (size_t)r * DMODEL;
  for (int d = tid; d < DMODEL; d += 256) yr[d] = (xr[d] - mean) * rst * g[d] + be[d];
}

__global__ void k_latent(const float* __restrict__ hid, const float* __restrict__ lw,
                         float* __restrict__ embeds, int S, int tok, int n) {
  int b = blockIdx.x;
  float w[3];
  float mx = -1e30f;
  for (int i = 0; i < n; ++i) { w[i] = lw[3 - n + i]; mx = fmaxf(mx, w[i]); }
  float sum = 0.f;
  for (int i = 0; i < n; ++i) { w[i] = __expf(w[i] - mx); sum += w[i]; }
  float inv = 1.0f / sum;
  float* dst = embeds + ((size_t)(b * TSEQ + tok)) * DMODEL;
  for (int d = threadIdx.x; d < DMODEL; d += 256) {
    float v = 0.f;
    for (int i = 0; i < n; ++i)
      v += w[i] * inv * hid[((size_t)(b * S + tok - n + i)) * DMODEL + d];
    dst[d] = v;
  }
}

__global__ void k_copy(float* __restrict__ dst, const float* __restrict__ src,
                       int S, int r0, int nr) {
  int i = blockIdx.x;
  int b = i / nr, rr = r0 + i % nr;
  const float* s = src + ((size_t)(b * S + rr)) * DMODEL;
  float* d = dst + ((size_t)(b * TSEQ + rr)) * DMODEL;
  for (int c = threadIdx.x; c < DMODEL; c += 256) d[c] = s[c];
}

// ---------- WMMA GEMM, TDM-fed double-buffered B operand ----------
// C[M,N] = act(A[M,K] @ Bt_g^T + bias) (+ residual); Bt_g is [N,K] bf16.
template <bool GELU, bool RES>
__global__ void k_gemm(const float* __restrict__ A, const __bf16* __restrict__ Bt_g,
                       const float* __restrict__ bias, const float* __restrict__ res,
                       float* __restrict__ C, int M, int N, int K) {
  __shared__ __bf16 As[64][40];       // row stride 80B (16B multiple)
  __shared__ __bf16 Bt[2][64][32];    // [buf][n][k]; TDM fills compact rows (64B)
  int tid = threadIdx.x;
  int wave = tid >> 5, lane = tid & 31;
  int m0 = blockIdx.y * 64, n0 = blockIdx.x * 64;
  int wm = (wave & 3) * 16;
  int wn = (wave >> 2) * 32;
  int col = lane & 15, half = lane >> 4;
  v8f acc0 = {}, acc1 = {};

  // prologue: DMA first B tile (rows n0..n0+63, k 0..31) into buffer 0
  if (wave == 0)
    tdm_load_2d_bf16(Bt_g + (size_t)n0 * K, &Bt[0][0][0],
                     (unsigned)K, (unsigned)N, 32u, 64u, (unsigned long long)K);

  int ib = 0;
  for (int kc = 0; kc < K; kc += 32, ib ^= 1) {
    { // stage A: 64 x 32 fp32 -> bf16 (activations need conversion; TDM can't)
      int mi = tid >> 2, ks = (tid & 3) * 8;
      int rg = m0 + mi;
      float4 f0 = make_float4(0, 0, 0, 0), f1 = f0;
      if (rg < M) {
        const float* ap = A + (size_t)rg * K + kc + ks;
        f0 = *(const float4*)ap; f1 = *(const float4*)(ap + 4);
      }
      __bf16* d = &As[mi][ks];
      d[0] = (__bf16)f0.x; d[1] = (__bf16)f0.y; d[2] = (__bf16)f0.z; d[3] = (__bf16)f0.w;
      d[4] = (__bf16)f1.x; d[5] = (__bf16)f1.y; d[6] = (__bf16)f1.z; d[7] = (__bf16)f1.w;
    }
    if (kc + 32 < K)  // prefetch next A K-tile -> global_prefetch_b8
      __builtin_prefetch(A + (size_t)(m0 + (tid >> 2)) * K + kc + 32, 0, 1);

    if (wave == 0) {
      if (kc + 32 < K) {  // DMA next B tile into other buffer, then wait current
        tdm_load_2d_bf16(Bt_g + (size_t)n0 * K + (kc + 32), &Bt[ib ^ 1][0][0],
                         (unsigned)K, (unsigned)N, 32u, 64u, (unsigned long long)K);
        __builtin_amdgcn_s_wait_tensorcnt(1);
      } else {
        __builtin_amdgcn_s_wait_tensorcnt(0);
      }
    }
    __syncthreads();

    v8bf alo = *(const v8bf*)&As[wm + col][half * 16];
    v8bf ahi = *(const v8bf*)&As[wm + col][half * 16 + 8];
    v16bf af = mk16(alo, ahi);
    v16bf bf0 = *(const v16bf*)&Bt[ib][wn + col][half * 16];
    v16bf bf1 = *(const v16bf*)&Bt[ib][wn + 16 + col][half * 16];
    acc0 = wmma_bf16(af, bf0, acc0);
    acc1 = wmma_bf16(af, bf1, acc1);
    __syncthreads();
  }

  int cn0 = n0 + wn + col, cn1 = cn0 + 16;
#pragma unroll
  for (int r = 0; r < 8; ++r) {
    int cm = m0 + wm + half * 8 + r;
    if (cm < M) {
      float v0 = acc0[r] + bias[cn0];
      float v1 = acc1[r] + bias[cn1];
      if (GELU) { v0 = gelu_tanh(v0); v1 = gelu_tanh(v1); }
      if (RES) { v0 += res[(size_t)cm * N + cn0]; v1 += res[(size_t)cm * N + cn1]; }
      C[(size_t)cm * N + cn0] = v0;
      C[(size_t)cm * N + cn1] = v1;
    }
  }
}

// ---------- WMMA flash attention (causal), 1 wave per (b,h,16-query tile) ----------
__global__ void k_attn(const float* __restrict__ qkv, float* __restrict__ ctx, int S) {
  __shared__ __bf16 Pbuf[16][32];
  __shared__ __bf16 Vt[64][32];
  int qt = blockIdx.x, bh = blockIdx.y;
  int b = bh >> 4, h = bh & 15;
  int lane = threadIdx.x, col = lane & 15, half = lane >> 4;
  int q0 = qt * 16;

  v16bf qf[2];
#pragma unroll
  for (int c = 0; c < 2; ++c) {
    const float* qp = qkv + ((size_t)(b * S + q0 + col)) * 3072 + h * 64 + c * 32;
#pragma unroll
    for (int i = 0; i < 8; ++i) {
      qf[c][i]     = (__bf16)qp[half * 16 + i];
      qf[c][i + 8] = (__bf16)qp[half * 16 + 8 + i];
    }
  }

  float mrun[8], lrun[8];
  v8f o[4] = {};
#pragma unroll
  for (int r = 0; r < 8; ++r) { mrun[r] = -3e38f; lrun[r] = 0.f; }

  int nkeys = min(S, q0 + 16);
  int kb = (nkeys + 31) >> 5;
  for (int t = 0; t < kb; ++t) {
    int ks = t * 32;
    {
      int key = ks + lane; if (key > S - 1) key = S - 1;
      const float* vp = qkv + ((size_t)(b * S + key)) * 3072 + 2048 + h * 64;
      for (int d = 0; d < 64; ++d) Vt[d][lane] = (__bf16)vp[d];
    }
    v8f s0 = {}, s1 = {};
#pragma unroll
    for (int j = 0; j < 2; ++j) {
      int key = ks + j * 16 + col; if (key > S - 1) key = S - 1;
      const float* kp = qkv + ((size_t)(b * S + key)) * 3072 + 1024 + h * 64;
#pragma unroll
      for (int c = 0; c < 2; ++c) {
        v16bf bf;
        const float* kpc = kp + c * 32 + half * 16;
#pragma unroll
        for (int e = 0; e < 16; ++e) bf[e] = (__bf16)kpc[e];
        if (j == 0) s0 = wmma_bf16(qf[c], bf, s0);
        else        s1 = wmma_bf16(qf[c], bf, s1);
      }
    }
    float p0[8], p1[8];
#pragma unroll
    for (int r = 0; r < 8; ++r) {
      int qrow = q0 + half * 8 + r;
      float v0 = s0[r] * 0.125f; if (ks + col > qrow)      v0 = -1e9f;
      float v1 = s1[r] * 0.125f; if (ks + 16 + col > qrow) v1 = -1e9f;
      float tm = rmax16(fmaxf(v0, v1));
      float mn = fmaxf(mrun[r], tm);
      float sc = __expf(mrun[r] - mn);
      float e0 = __expf(v0 - mn), e1 = __expf(v1 - mn);
      float rs = rsum16(e0 + e1);
      lrun[r] = lrun[r] * sc + rs;
      mrun[r] = mn;
#pragma unroll
      for (int n = 0; n < 4; ++n) o[n][r] *= sc;
      p0[r] = e0; p1[r] = e1;
    }
#pragma unroll
    for (int r = 0; r < 8; ++r) {
      Pbuf[half * 8 + r][col]      = (__bf16)p0[r];
      Pbuf[half * 8 + r][16 + col] = (__bf16)p1[r];
    }
    __syncthreads();
    v8bf plo = *(const v8bf*)&Pbuf[col][half * 16];
    v8bf phi = *(const v8bf*)&Pbuf[col][half * 16 + 8];
    v16bf pf = mk16(plo, phi);
#pragma unroll
    for (int n = 0; n < 4; ++n) {
      v16bf vf = *(const v16bf*)&Vt[n * 16 + col][half * 16];
      o[n] = wmma_bf16(pf, vf, o[n]);
    }
    __syncthreads();
  }
#pragma unroll
  for (int n = 0; n < 4; ++n)
#pragma unroll
    for (int r = 0; r < 8; ++r) {
      int qrow = q0 + half * 8 + r;
      if (qrow < S)
        ctx[((size_t)(b * S + qrow)) * DMODEL + h * 64 + n * 16 + col] =
            o[n][r] / lrun[r];
    }
}

// ---------- fused logits + log-softmax + NLL over tied embedding ----------
__global__ void k_nll(const float* __restrict__ hid, const float* __restrict__ emb,
                      const int* __restrict__ labels, float* __restrict__ out,
                      float invCount) {
  __shared__ __bf16 Ah[16][1032];
  __shared__ float wmx[8][16], wls[8][16], lablog[16];
  __shared__ int labv[16];
  int b = blockIdx.y, t0 = blockIdx.x * 16;
  int tid = threadIdx.x, wave = tid >> 5, lane = tid & 31;
  int col = lane & 15, half = lane >> 4;

  {
    int mi = tid >> 4, ks = (tid & 15) * 64;
    const float* hp = hid + ((size_t)(b * TSEQ + t0 + mi)) * DMODEL + ks;
    for (int i = 0; i < 64; i += 4) {
      float4 f = *(const float4*)(hp + i);
      Ah[mi][ks + i]     = (__bf16)f.x; Ah[mi][ks + i + 1] = (__bf16)f.y;
      Ah[mi][ks + i + 2] = (__bf16)f.z; Ah[mi][ks + i + 3] = (__bf16)f.w;
    }
  }
  if (tid < 16) {
    int t = t0 + tid;
    labv[tid] = (t < TSEQ - 1) ? labels[b * TSEQ + t + 1] : -1;
    lablog[tid] = 0.f;
  }
  __syncthreads();

  float mrun[8], lrun[8];
#pragma unroll
  for (int r = 0; r < 8; ++r) { mrun[r] = -3e38f; lrun[r] = 0.f; }

  for (int vt = wave; vt < VOCAB / 16; vt += 8) {
    int v0 = vt * 16;
    v8f acc = {};
    const float* ep0 = emb + (size_t)(v0 + col) * DMODEL + half * 16;
    for (int kc = 0; kc < DMODEL; kc += 32) {
      v8bf alo = *(const v8bf*)&Ah[col][kc + half * 16];
      v8bf ahi = *(const v8bf*)&Ah[col][kc + half * 16 + 8];
      v16bf af = mk16(alo, ahi);
      const float* ep = ep0 + kc;
      v16bf bf;
#pragma unroll
      for (int e = 0; e < 16; ++e) bf[e] = (__bf16)ep[e];
      acc = wmma_bf16(af, bf, acc);
    }
#pragma unroll
    for (int r = 0; r < 8; ++r) {
      int m = half * 8 + r;
      float v = acc[r];
      if (v0 + col == labv[m]) lablog[m] = v;
      float tm = rmax16(v);
      float mn = fmaxf(mrun[r], tm);
      float rs = rsum16(__expf(v - mn));
      lrun[r] = lrun[r] * __expf(mrun[r] - mn) + rs;
      mrun[r] = mn;
    }
  }
  if (col == 0) {
#pragma unroll
    for (int r = 0; r < 8; ++r) {
      int m = half * 8 + r;
      wmx[wave][m] = mrun[r];
      wls[wave][m] = lrun[r];
    }
  }
  __syncthreads();
  if (tid < 16) {
    int t = t0 + tid;
    if (t < TSEQ - 1) {
      float M = -3e38f;
      for (int w = 0; w < 8; ++w) M = fmaxf(M, wmx[w][tid]);
      float Ss = 0.f;
      for (int w = 0; w < 8; ++w) Ss += wls[w][tid] * __expf(wmx[w][tid] - M);
      float nll = M + __logf(Ss) - lablog[tid];
      atomicAdd(out, nll * invCount);
    }
  }
}

// ---------- host orchestration ----------
extern "C" void kernel_launch(void* const* d_in, const int* in_sizes, int n_in,
                              void* d_out, int out_size, void* d_ws, size_t ws_size,
                              hipStream_t stream) {
  const int*   ids    = (const int*)d_in[0];
  const int*   labels = (const int*)d_in[2];
  const float* emb    = (const float*)d_in[4];
  const float* Wpos   = (const float*)d_in[5];
  const float* Wqkv   = (const float*)d_in[6];
  const float* bqkv   = (const float*)d_in[7];
  const float* Aq     = (const float*)d_in[8];
  const float* Bq     = (const float*)d_in[9];
  const float* Wo     = (const float*)d_in[10];
  const float* bo     = (const float*)d_in[11];
  const float* Ao     = (const float*)d_in[12];
  const float* Bo     = (const float*)d_in[13];
  const float* W1     = (const float*)d_in[14];
  const float* b1     = (const float*)d_in[15];
  const float* W2     = (const float*)d_in[16];
  const float* b2     = (const float*)d_in[17];
  const float* ln1s   = (const float*)d_in[18];
  const float* ln1b   = (const float*)d_in[19];
  const float* ln2s   = (const float*)d_in[20];
  const float* ln2b   = (const float*)d_in[21];
  const float* lnfs   = (const float*)d_in[22];
  const float* lnfb   = (const float*)d_in[23];
  const float* latw   = (const float*)d_in[24];

  char* ws = (char*)d_ws;
  size_t off = 0;
  auto alloc = [&](size_t bytes) {
    void* p = ws + off;
    off += (bytes + 255) & ~(size_t)255;
    return p;
  };
  __bf16* WqkvT = (__bf16*)alloc((size_t)3 * DMODEL * DMODEL * 2);  // [3072][1024]
  __bf16* WoT   = (__bf16*)alloc((size_t)DMODEL * DMODEL * 2);      // [1024][1024]
  __bf16* W1T   = (__bf16*)alloc((size_t)DFF_ * DMODEL * 2);        // [4096][1024]
  __bf16* W2T   = (__bf16*)alloc((size_t)DMODEL * DFF_ * 2);        // [1024][4096]
  float* embeds = (float*)alloc((size_t)2 * TSEQ * DMODEL * 4);
  float* h      = (float*)alloc((size_t)2 * TSEQ * DMODEL * 4);
  float* a      = (float*)alloc((size_t)2 * TSEQ * DMODEL * 4);
  float* qkvb   = (float*)alloc((size_t)2 * TSEQ * 3 * DMODEL * 4);
  float* ctx    = (float*)alloc((size_t)2 * TSEQ * DMODEL * 4);
  float* ff     = (float*)alloc((size_t)2 * TSEQ * DFF_ * 4);
  float* hidp   = (float*)alloc((size_t)2 * TSEQ * DMODEL * 4);
  float* hall   = (float*)alloc((size_t)2 * TSEQ * DMODEL * 4);

  hipMemsetAsync(d_out, 0, sizeof(float), stream);

  long tq = (long)DMODEL * 3 * DMODEL;
  k_merge_t<<<(int)((tq + 255) / 256), 256, 0, stream>>>(Wqkv, Aq, Bq, WqkvT,
                                                         DMODEL, 3 * DMODEL, tq);
  long to = (long)DMODEL * DMODEL;
  k_merge_t<<<(int)((to + 255) / 256), 256, 0, stream>>>(Wo, Ao, Bo, WoT,
                                                         DMODEL, DMODEL, to);
  long t1 = (long)DMODEL * DFF_;
  k_convT<<<(int)((t1 + 255) / 256), 256, 0, stream>>>(W1, W1T, DMODEL, DFF_, t1);
  k_convT<<<(int)((t1 + 255) / 256), 256, 0, stream>>>(W2, W2T, DFF_, DMODEL, t1);
  k_embed<<<2 * TSEQ, 256, 0, stream>>>(ids, emb, embeds);

  const int Ss[5] = {512, 513, 514, 515, 1024};
  for (int p = 0; p < 5; ++p) {
    int S = Ss[p], M = 2 * S;
    k_addpos<<<M, 256, 0, stream>>>(embeds, Wpos, h, S);
    k_ln<<<M, 256, 0, stream>>>(h, ln1s, ln1b, a);
    dim3 g1(3 * DMODEL / 64, (M + 63) / 64);
    k_gemm<false, false><<<g1, 256, 0, stream>>>(a, WqkvT, bqkv, nullptr, qkvb,
                                                 M, 3 * DMODEL, DMODEL);
    dim3 ga((S + 15) / 16, 2 * NHEAD);
    k_attn<<<ga, 32, 0, stream>>>(qkvb, ctx, S);
    dim3 g2(DMODEL / 64, (M + 63) / 64);
    k_gemm<false, true><<<g2, 256, 0, stream>>>(ctx, WoT, bo, h, h, M, DMODEL, DMODEL);
    k_ln<<<M, 256, 0, stream>>>(h, ln2s, ln2b, a);
    dim3 g3(DFF_ / 64, (M + 63) / 64);
    k_gemm<true, false><<<g3, 256, 0, stream>>>(a, W1T, b1, nullptr, ff, M, DFF_, DMODEL);
    dim3 g4(DMODEL / 64, (M + 63) / 64);
    k_gemm<false, true><<<g4, 256, 0, stream>>>(ff, W2T, b2, h, h, M, DMODEL, DFF_);
    k_ln<<<M, 256, 0, stream>>>(h, lnfs, lnfb, hidp);

    int r0 = (p == 0) ? 0 : (511 + p);
    int nr = (p == 0) ? 512 : (p < 4 ? 1 : (TSEQ - 515));
    k_copy<<<2 * nr, 256, 0, stream>>>(hall, hidp, S, r0, nr);
    if (p < 4) {
      int tok = 512 + p;
      int n = (p == 0) ? 3 : 1;
      k_latent<<<2, 256, 0, stream>>>(hidp, latw, embeds, S, tok, n);
    }
  }
  dim3 gn(TSEQ / 16, 2);
  k_nll<<<gn, 256, 0, stream>>>(hall, emb, labels, (float*)d_out,
                                1.0f / (2.0f * (TSEQ - 1)));
}
